// MultiHeadAttention_62861141344769
// MI455X (gfx1250) — compile-verified, matrix operand order checked by reading, not run on hardware
//
#include <hip/hip_runtime.h>

#define B_  2
#define N_  2048
#define D_  1024
#define H_  16
#define DK_ 64

typedef __attribute__((ext_vector_type(16))) __bf16       v16bf;
typedef __attribute__((ext_vector_type(8)))  float        v8f;
typedef __attribute__((ext_vector_type(4)))  unsigned int v4u;

union Frag {
    v16bf bf;
    v4u   u[2];
    unsigned short us[16];
};

__device__ __forceinline__ unsigned short f2bf(float f) {
    unsigned int u = __float_as_uint(f);
    u += 0x7fffu + ((u >> 16) & 1u);   // round-to-nearest-even
    return (unsigned short)(u >> 16);
}

// Load a 16-row x 32-K bf16 WMMA fragment (A or B operand) from a row-major
// matrix with leading dimension `ld` (elements). Per the CDNA5 16-bit A/B
// layout: lane = row (mod 16), halves 0-7 at K = k0 + sel*8, halves 8-15 at
// K = k0 + 16 + sel*8, where sel = lane>>4. Two b128 loads per lane.
__device__ __forceinline__ v16bf load_frag(const unsigned short* base, int ld,
                                           int row0, int k0, int lane) {
    int r = lane & 15, sel = lane >> 4;
    const unsigned short* p = base + (size_t)(row0 + r) * ld + k0 + sel * 8;
    Frag f;
    f.u[0] = *(const v4u*)(p);
    f.u[1] = *(const v4u*)(p + 16);
    return f.bf;
}

__device__ __forceinline__ v8f wmma_bf16(v16bf a, v16bf b, v8f c) {
    return __builtin_amdgcn_wmma_f32_16x16x32_bf16(false, a, false, b,
                                                   (short)0, c, false, false);
}

// ---------------------------------------------------------------------------
// fp32 -> bf16 conversion
// ---------------------------------------------------------------------------
__global__ void cvt_bf16_kernel(const float* __restrict__ src,
                                unsigned short* __restrict__ dst, int n) {
    int i = blockIdx.x * blockDim.x + threadIdx.x;
    if (i < n) dst[i] = f2bf(src[i]);
}

// ---------------------------------------------------------------------------
// GEMM: O[m,c] = sum_k A[m,k] * W[c,k]   (M = B*N = 4096, K = C = D = 1024)
// One wave computes a 32x64 output tile (2x4 WMMA subtiles) with register
// double-buffering of the 6 fragments per K-step: 8 WMMAs per 6 b128-pair
// loads, ~160 live VGPRs (no spills; 256 is the hard budget).
// MODE 0: store bf16 into Q/K layout (B,H,N,DK)
// MODE 1: store bf16 into Vt layout  (B,H,DK,N)
// MODE 2: store fp32 row-major with bias (final output projection)
// MODE 3: like MODE 0 but scaled by 1/sqrt(DK) (Q projection)
// ---------------------------------------------------------------------------
template <int MODE>
__global__ void __launch_bounds__(128, 1)
gemm_wmma_kernel(const unsigned short* __restrict__ A,
                 const unsigned short* __restrict__ W,
                 void* __restrict__ dst,
                 const float* __restrict__ bias) {
    int wave = (int)((blockIdx.x * blockDim.x + threadIdx.x) >> 5);
    int lane = threadIdx.x & 31;
    const int NT = D_ / 64;               // 16 column tiles
    int mt = wave / NT, nt = wave % NT;
    int m0 = mt * 32, n0 = nt * 64;

    v8f acc[2][4] = {};
    v16bf af[2], bf_[4], afn[2], bfn[4];
#pragma unroll
    for (int i = 0; i < 2; ++i) af[i]  = load_frag(A, D_, m0 + i * 16, 0, lane);
#pragma unroll
    for (int j = 0; j < 4; ++j) bf_[j] = load_frag(W, D_, n0 + j * 16, 0, lane);

    for (int kk = 0; kk < D_; kk += 32) {
        int kn = (kk + 32 < D_) ? kk + 32 : 0;   // safe dummy on last iter
        // prefetch next K-step fragments; they complete under this step's WMMAs
#pragma unroll
        for (int i = 0; i < 2; ++i) afn[i] = load_frag(A, D_, m0 + i * 16, kn, lane);
#pragma unroll
        for (int j = 0; j < 4; ++j) bfn[j] = load_frag(W, D_, n0 + j * 16, kn, lane);
#pragma unroll
        for (int i = 0; i < 2; ++i)
#pragma unroll
            for (int j = 0; j < 4; ++j)
                acc[i][j] = wmma_bf16(af[i], bf_[j], acc[i][j]);
#pragma unroll
        for (int i = 0; i < 2; ++i) af[i]  = afn[i];
#pragma unroll
        for (int j = 0; j < 4; ++j) bf_[j] = bfn[j];
    }

    // C/D layout: VGPR r -> row r (lanes 0-15) / row r+8 (lanes 16-31),
    // column = lane & 15.
    int col = lane & 15, sel = lane >> 4;
#pragma unroll
    for (int i = 0; i < 2; ++i)
#pragma unroll
        for (int j = 0; j < 4; ++j)
#pragma unroll
            for (int r = 0; r < 8; ++r) {
                int m = m0 + i * 16 + sel * 8 + r;
                int c = n0 + j * 16 + col;
                float v = acc[i][j][r];
                if (MODE == 2) {
                    ((float*)dst)[(size_t)m * D_ + c] = v + bias[c];
                } else {
                    if (MODE == 3) v *= 0.125f;           // 1/sqrt(64)
                    int b = m >> 11, n = m & (N_ - 1);    // N_ = 2048
                    int h = c >> 6,  d = c & (DK_ - 1);
                    size_t idx = (MODE == 1)
                        ? (((size_t)(b * H_ + h) * DK_ + d) * N_ + n)
                        : (((size_t)(b * H_ + h) * N_ + n) * DK_ + d);
                    ((unsigned short*)dst)[idx] = f2bf(v);
                }
            }
}

// ---------------------------------------------------------------------------
// Causal flash attention, one wave per (b, h, 32-query tile).
// Computes S^T = K_chunk @ Q^T so that softmaxed probabilities land directly
// in the B-fragment layout for out^T = Vt_tile @ P^T (no LDS bounce, no
// cross-lane data movement except one shfl_xor(16) per reduction).
// Two query sub-tiles share each K/V chunk load (16 WMMAs per 8 fragment
// loads). V-fragment loads are issued under the S^T wait + softmax VALU;
// next chunk's K/V streams are warmed with global_prefetch_b8.
// Writes attn (B, N, D) in bf16.  (Q pre-scaled by 1/sqrt(DK) in projection.)
// ---------------------------------------------------------------------------
__global__ void __launch_bounds__(128, 1)
attn_kernel(const unsigned short* __restrict__ Q,
            const unsigned short* __restrict__ K,
            const unsigned short* __restrict__ Vt,
            unsigned short* __restrict__ attn) {
    int wave = (int)((blockIdx.x * blockDim.x + threadIdx.x) >> 5);
    int lane = threadIdx.x & 31;
    const int QT = N_ / 32;              // 64 query tiles per (b,h)
    int qt = wave & (QT - 1);
    int bh = wave / QT;                  // 0..31
    int q0 = qt * 32;

    const unsigned short* Qb = Q  + (size_t)bh * N_ * DK_;
    const unsigned short* Kb = K  + (size_t)bh * N_ * DK_;
    const unsigned short* Vb = Vt + (size_t)bh * DK_ * N_;

    // Q fragments: B operand of S^T. qf[s][h]: query sub-tile s, d-half h.
    v16bf qf[2][2];
#pragma unroll
    for (int s = 0; s < 2; ++s) {
        qf[s][0] = load_frag(Qb, DK_, q0 + s * 16, 0,  lane);
        qf[s][1] = load_frag(Qb, DK_, q0 + s * 16, 32, lane);
    }

    v8f acc[4][2] = {};                  // out^T: [d-subtile][q-subtile]
    float m_run[2] = {-INFINITY, -INFINITY};
    float l_run[2] = {0.0f, 0.0f};
    int col = lane & 15, sel = lane >> 4;

    int nch = (q0 + 31) / 32 + 1;        // 32-key chunks (causal bound)

    for (int kc = 0; kc < nch; ++kc) {
        int kb = kc * 32;

        // K fragments for this chunk: kcur[t*2+h], key sub-tile t, d-half h
        v16bf kcur[4];
#pragma unroll
        for (int t = 0; t < 2; ++t) {
            kcur[t * 2 + 0] = load_frag(Kb, DK_, kb + t * 16, 0,  lane);
            kcur[t * 2 + 1] = load_frag(Kb, DK_, kb + t * 16, 32, lane);
        }

        // S^T tiles: st[t][s] = K subtile t x Q subtile s (keys x queries)
        v8f st[2][2];
#pragma unroll
        for (int t = 0; t < 2; ++t)
#pragma unroll
            for (int s = 0; s < 2; ++s) {
                v8f z = {};
                z = wmma_bf16(kcur[t * 2 + 0], qf[s][0], z);
                st[t][s] = wmma_bf16(kcur[t * 2 + 1], qf[s][1], z);
            }

        // issue V fragment loads now; they complete under the softmax VALU
        v16bf vf[4];
#pragma unroll
        for (int dt = 0; dt < 4; ++dt)
            vf[dt] = load_frag(Vb, N_, dt * 16, kb, lane);

        // warm next chunk's K/V streams (global_prefetch_b8, no counter cost)
        int kbn = (kc + 1 < nch) ? kb + 32 : 0;
        __builtin_prefetch(Kb + (size_t)(kbn + lane) * DK_, 0, 0);
        __builtin_prefetch(Vb + (size_t)lane * N_ + kbn, 0, 0);
        __builtin_prefetch(Vb + (size_t)(32 + lane) * N_ + kbn, 0, 0);

        // online softmax per query sub-tile (column q split across lane^16)
        float alpha[2];
        Frag pf[2];
#pragma unroll
        for (int s = 0; s < 2; ++s) {
            int qg = q0 + s * 16 + col;
            float sv[16];
#pragma unroll
            for (int i = 0; i < 8; ++i) {
                sv[i]     = st[0][s][i];
                sv[8 + i] = st[1][s][i];
            }
            if (kb + 31 > q0 + s * 16) {          // diagonal chunk: mask
#pragma unroll
                for (int i = 0; i < 8; ++i) {
                    if (kb + sel * 8 + i      > qg) sv[i]     = -INFINITY;
                    if (kb + 16 + sel * 8 + i > qg) sv[8 + i] = -INFINITY;
                }
            }
            float mc = sv[0];
#pragma unroll
            for (int i = 1; i < 16; ++i) mc = fmaxf(mc, sv[i]);
            mc = fmaxf(mc, __shfl_xor(mc, 16, 32));
            float m_new = fmaxf(m_run[s], mc);
            alpha[s] = __expf(m_run[s] - m_new);
            float psum = 0.0f;
#pragma unroll
            for (int i = 0; i < 16; ++i) {
                float p = __expf(sv[i] - m_new);
                psum += p;
                pf[s].us[i] = f2bf(p);    // lands exactly in B-fragment order
            }
            psum += __shfl_xor(psum, 16, 32);
            l_run[s] = l_run[s] * alpha[s] + psum;
            m_run[s] = m_new;
        }

        // rescale accumulators and accumulate P @ V
#pragma unroll
        for (int dt = 0; dt < 4; ++dt)
#pragma unroll
            for (int s = 0; s < 2; ++s) {
#pragma unroll
                for (int r = 0; r < 8; ++r) acc[dt][s][r] *= alpha[s];
                acc[dt][s] = wmma_bf16(vf[dt], pf[s].bf, acc[dt][s]);
            }
    }

    int b = bh >> 4, h = bh & 15;
#pragma unroll
    for (int s = 0; s < 2; ++s) {
        float inv_l = 1.0f / l_run[s];
        int qg = q0 + s * 16 + col;
#pragma unroll
        for (int dt = 0; dt < 4; ++dt)
#pragma unroll
            for (int r = 0; r < 8; ++r) {
                int d = dt * 16 + sel * 8 + r;
                size_t idx = ((size_t)b * N_ + qg) * D_ + h * DK_ + d;
                attn[idx] = f2bf(acc[dt][s][r] * inv_l);
            }
    }
}

// ---------------------------------------------------------------------------
extern "C" void kernel_launch(void* const* d_in, const int* in_sizes, int n_in,
                              void* d_out, int out_size, void* d_ws, size_t ws_size,
                              hipStream_t stream) {
    const float* x  = (const float*)d_in[0];
    // d_in[1] = mask (causal; implemented analytically, unused)
    const float* Wq = (const float*)d_in[2];
    const float* Wk = (const float*)d_in[3];
    const float* Wv = (const float*)d_in[4];
    const float* Wo = (const float*)d_in[5];
    const float* bo = (const float*)d_in[6];
    float* out = (float*)d_out;

    const size_t XE = (size_t)B_ * N_ * D_;   // 4,194,304
    const size_t WE = (size_t)D_ * D_;        // 1,048,576
    unsigned short* xb   = (unsigned short*)d_ws;
    unsigned short* wqb  = xb  + XE;
    unsigned short* wkb  = wqb + WE;
    unsigned short* wvb  = wkb + WE;
    unsigned short* wob  = wvb + WE;
    unsigned short* Qb   = wob + WE;
    unsigned short* Kb   = Qb  + XE;
    unsigned short* Vtb  = Kb  + XE;
    unsigned short* atnb = Vtb + XE;          // total 48 MB

    // fp32 -> bf16 conversions
    cvt_bf16_kernel<<<(int)((XE + 255) / 256), 256, 0, stream>>>(x,  xb,  (int)XE);
    cvt_bf16_kernel<<<(int)((WE + 255) / 256), 256, 0, stream>>>(Wq, wqb, (int)WE);
    cvt_bf16_kernel<<<(int)((WE + 255) / 256), 256, 0, stream>>>(Wk, wkb, (int)WE);
    cvt_bf16_kernel<<<(int)((WE + 255) / 256), 256, 0, stream>>>(Wv, wvb, (int)WE);
    cvt_bf16_kernel<<<(int)((WE + 255) / 256), 256, 0, stream>>>(Wo, wob, (int)WE);

    // projections: 2048 waves (32x64 tiles), 4 waves per 128-thread block
    dim3 gthr(128);
    gemm_wmma_kernel<3><<<dim3(512), gthr, 0, stream>>>(xb, wqb, (void*)Qb,  nullptr);
    gemm_wmma_kernel<0><<<dim3(512), gthr, 0, stream>>>(xb, wkb, (void*)Kb,  nullptr);
    gemm_wmma_kernel<1><<<dim3(512), gthr, 0, stream>>>(xb, wvb, (void*)Vtb, nullptr);

    // flash attention: 2048 waves (one per (b,h,32-query tile))
    attn_kernel<<<dim3(512), gthr, 0, stream>>>(Qb, Kb, Vtb, atnb);

    // output projection + bias, fp32 out
    gemm_wmma_kernel<2><<<dim3(512), gthr, 0, stream>>>(atnb, wob, (void*)out, bo);
}
